// MultiheadAttention_71193377899400
// MI455X (gfx1250) — matrix operand order repaired and checked by reading, not compiled, not run on hardware
//
#include <hip/hip_runtime.h>
#include <hip/hip_bf16.h>

// ---------------------------------------------------------------------------
// Swin window attention for MI455X (gfx1250), bf16 WMMA with f32 accumulate.
// One workgroup per window (2048 windows), 256 threads = 8 wave32 waves.
// A prep kernel stages bf16 weights + pre-gathered rel-pos bias in d_ws
// (weights/bias are shared by all windows -> L2-resident, no per-window cvt).
// ---------------------------------------------------------------------------

typedef __bf16 bf16;
typedef bf16  v16bf __attribute__((ext_vector_type(16)));
typedef bf16  v8bf  __attribute__((ext_vector_type(8)));
typedef float v8f   __attribute__((ext_vector_type(8)));
typedef float v4f   __attribute__((ext_vector_type(4)));

#define LDX 392      // xn / attn-out LDS row stride (bf16 elems), padded (conflict-free b128)
#define LDQ 1160     // qkv LDS row stride (bf16 elems), padded
#define LDP 72       // per-wave P-scratch row stride
#define SMEM_BYTES (64*LDX*2 + 64*LDQ*2 + 8*16*LDP*2)   // 50176 + 148480 + 18432 = 217088

// Workspace layout (bytes)
#define NQKV  (1152*384)
#define NOUT  (384*384)
#define NBIAS (12*64*64)
#define WS_OFF_WQKV 0
#define WS_OFF_WOUT (NQKV*2)                 // 884736
#define WS_OFF_BIAS (NQKV*2 + NOUT*2)        // 1179648
#define WS_REQ      (NQKV*2 + NOUT*2 + NBIAS*4)  // 1376256

static __device__ __forceinline__ v8f wmma_bf16(v16bf a, v16bf b, v8f c) {
  // v_wmma_f32_16x16x32_bf16: D = A(16x32) * B(32x16) + C(16x16 f32)
  return __builtin_amdgcn_wmma_f32_16x16x32_bf16(false, a, false, b, (short)0, c, false, false);
}

// A-matrix 16x32 bf16 fragment from row-major source.
// Layout: lanes 0-15 hold row (lane), K = 0..15; lanes 16-31 same rows, K = 16..31 (matches B loaders).
static __device__ __forceinline__ v16bf load_a_frag(const bf16* base, int row0, int col0, int ld, int lane) {
  const int r  = lane & 15;
  const int hi = lane >> 4;
  const bf16* p = base + (row0 + r) * ld + col0 + hi * 16;
  v8bf lo = *(const v8bf*)p;          // ds_load_b128
  v8bf hh = *(const v8bf*)(p + 8);    // ds_load_b128
  return __builtin_shufflevector(lo, hh, 0,1,2,3,4,5,6,7,8,9,10,11,12,13,14,15);
}

// B-matrix 32x16 bf16 fragment from a row-major B^T source (LDS or global).
// Lane n(0-15) holds column n, K = col0+0..15; lanes 16-31: K = col0+16..31.
static __device__ __forceinline__ v16bf load_bt_frag(const bf16* base, int row0, int col0, int ld, int lane) {
  const int n  = lane & 15;
  const int hi = lane >> 4;
  const bf16* p = base + (row0 + n) * ld + col0 + hi * 16;
  v8bf lo = *(const v8bf*)p;
  v8bf hh = *(const v8bf*)(p + 8);
  return __builtin_shufflevector(lo, hh, 0,1,2,3,4,5,6,7,8,9,10,11,12,13,14,15);
}

// Fallback: same as load_bt_frag but from f32 global weights (convert inline).
static __device__ __forceinline__ v16bf load_bt_frag_f32(const float* __restrict__ base,
                                                         int row0, int col0, int ld, int lane) {
  const int n  = lane & 15;
  const int hi = lane >> 4;
  const float* p = base + (row0 + n) * ld + col0 + hi * 16;
  v4f a = *(const v4f*)(p + 0);
  v4f b = *(const v4f*)(p + 4);
  v4f c = *(const v4f*)(p + 8);
  v4f d = *(const v4f*)(p + 12);
  v16bf f;
#pragma unroll
  for (int i = 0; i < 4; ++i) {
    f[i]      = (bf16)a[i];
    f[4 + i]  = (bf16)b[i];
    f[8 + i]  = (bf16)c[i];
    f[12 + i] = (bf16)d[i];
  }
  return f;
}

// B fragment where B^T is stored column-major, i.e. V[m][d] row-major: strided u16 loads.
static __device__ __forceinline__ v16bf load_v_frag(const bf16* vbase, int k0, int ld, int lane) {
  const int d  = lane & 15;
  const int hi = lane >> 4;
  v16bf f;
#pragma unroll
  for (int j = 0; j < 16; ++j)
    f[j] = vbase[(k0 + hi * 16 + j) * ld + d];
  return f;
}

// C/D 16x16 f32 tile -> bf16 LDS (row = vgpr + 8*laneHalf, col = lane&15).
static __device__ __forceinline__ void store_d_bf16(bf16* base, int row0, int col0, int ld,
                                                    int lane, v8f d, float scale) {
  const int c  = lane & 15;
  const int hi = lane >> 4;
#pragma unroll
  for (int g = 0; g < 8; ++g)
    base[(row0 + hi * 8 + g) * ld + col0 + c] = (bf16)(d[g] * scale);
}

// ---------------------------------------------------------------------------
// Prep kernel: f32 weights -> bf16 in ws; expand bias_table[rel_idx] -> [12][64][64]
// ---------------------------------------------------------------------------
__global__ __launch_bounds__(256) void prep_ws(
    const float* __restrict__ w_qkv, const float* __restrict__ w_out,
    const float* __restrict__ bias_table, const int* __restrict__ rel_idx,
    bf16* __restrict__ wqkv_bf, bf16* __restrict__ wout_bf, float* __restrict__ bias_pre) {
  const int i = blockIdx.x * 256 + threadIdx.x;
  if (i < NQKV) {
    wqkv_bf[i] = (bf16)w_qkv[i];
  } else if (i < NQKV + NOUT) {
    const int j = i - NQKV;
    wout_bf[j] = (bf16)w_out[j];
  } else if (i < NQKV + NOUT + NBIAS) {
    const int j  = i - NQKV - NOUT;   // j = h*4096 + n*64 + m
    const int h  = j >> 12;
    const int nm = j & 4095;
    bias_pre[j] = bias_table[rel_idx[nm] * 12 + h];
  }
}

extern __shared__ char smem_raw[];

template <bool PRE>
__global__ __launch_bounds__(256) void swin_window_attn(
    const float* __restrict__ x, const float* __restrict__ ln_g, const float* __restrict__ ln_b,
    const float* __restrict__ w_qkv, const float* __restrict__ w_out, const float* __restrict__ b_out,
    const float* __restrict__ bias_table, const int* __restrict__ rel_idx,
    const bf16* __restrict__ wqkv_bf, const bf16* __restrict__ wout_bf,
    const float* __restrict__ bias_pre,
    float* __restrict__ out) {
  bf16* xn  = (bf16*)smem_raw;                                  // [64][LDX]; later reused as attn out
  bf16* qkv = (bf16*)(smem_raw + 64 * LDX * 2);                 // [64][LDQ]: cols 0-383 Q(scaled), 384-767 K, 768-1151 V
  bf16* psc = (bf16*)(smem_raw + 64 * LDX * 2 + 64 * LDQ * 2);  // [8 waves][16][LDP]

  const int tid  = threadIdx.x;
  const int lane = tid & 31;
  const int wave = tid >> 5;
  const int win  = blockIdx.x;
  const size_t xbase = (size_t)win * 64 * 384;

  // ---------------- Phase 1: LayerNorm -> bf16 xn in LDS -------------------
  for (int n = wave; n < 64; n += 8) {
    float v[12];
    float s = 0.f, ss = 0.f;
#pragma unroll
    for (int i = 0; i < 12; ++i) {
      v[i] = x[xbase + n * 384 + lane + 32 * i];
      s += v[i];
    }
#pragma unroll
    for (int i = 0; i < 12; ++i) ss += v[i] * v[i];
#pragma unroll
    for (int m = 16; m >= 1; m >>= 1) {
      s  += __shfl_xor(s,  m, 32);
      ss += __shfl_xor(ss, m, 32);
    }
    const float mean = s * (1.f / 384.f);
    const float var  = ss * (1.f / 384.f) - mean * mean;
    const float inv  = rsqrtf(var + 1e-5f);
#pragma unroll
    for (int i = 0; i < 12; ++i) {
      const int c = lane + 32 * i;
      xn[n * LDX + c] = (bf16)((v[i] - mean) * inv * ln_g[c] + ln_b[c]);
    }
  }
  __syncthreads();

  // ---------------- Phase 2: QKV GEMM (64x384 x 384x1152) ------------------
  // qkv[n][e] = sum_c xn[n][c] * w_qkv[e][c]; w_qkv row-major [e][c] == B^T.
  const float qscale = 0.17677669529663687f;  // 1/sqrt(32), folded into Q
  for (int nt = wave; nt < 72; nt += 8) {
    const int e0 = nt * 16;
    v16bf bfr[12];
#pragma unroll
    for (int k = 0; k < 12; ++k)
      bfr[k] = PRE ? load_bt_frag(wqkv_bf, e0, k * 32, 384, lane)
                   : load_bt_frag_f32(w_qkv, e0, k * 32, 384, lane);
    const float sc = (e0 < 384) ? qscale : 1.0f;
    for (int mt = 0; mt < 4; ++mt) {
      v8f acc = {0.f, 0.f, 0.f, 0.f, 0.f, 0.f, 0.f, 0.f};
#pragma unroll
      for (int k = 0; k < 12; ++k) {
        v16bf a = load_a_frag(xn, mt * 16, k * 32, LDX, lane);
        acc = wmma_bf16(a, bfr[k], acc);
      }
      store_d_bf16(qkv, mt * 16, e0, LDQ, lane, acc, sc);
    }
  }
  __syncthreads();

  // ---------------- Phase 3: per-head attention ----------------------------
  bf16* pw = psc + wave * 16 * LDP;
  for (int h = wave; h < 12; h += 8) {
    const int qc = h * 32, kc = 384 + h * 32, vc = 768 + h * 32;
    v16bf kfr[4];          // K as B: B^T rows = tokens m, K-dim = d (row-major in qkv)
#pragma unroll
    for (int t = 0; t < 4; ++t)
      kfr[t] = load_bt_frag(qkv, t * 16, kc, LDQ, lane);
    v16bf vfr[2][2];       // V as B: B^T[d][m] = V[m][d] -> strided loads
#pragma unroll
    for (int dt = 0; dt < 2; ++dt)
#pragma unroll
      for (int kt = 0; kt < 2; ++kt)
        vfr[dt][kt] = load_v_frag(qkv + vc + dt * 16, kt * 32, LDQ, lane);

    for (int mt = 0; mt < 4; ++mt) {
      v16bf a = load_a_frag(qkv, mt * 16, qc, LDQ, lane);  // Q rows (pre-scaled)
      v8f sacc[4];
#pragma unroll
      for (int nt = 0; nt < 4; ++nt) {
        // Seed accumulator with relative-position bias: scores = Q*K^T*scale + bias
        v8f c;
#pragma unroll
        for (int g = 0; g < 8; ++g) {
          const int row = mt * 16 + (lane >> 4) * 8 + g;
          const int col = nt * 16 + (lane & 15);
          if (PRE) {
            c[g] = bias_pre[(h << 12) + row * 64 + col];   // coalesced over 16 lanes
          } else {
            const int idx = rel_idx[row * 64 + col];
            c[g] = bias_table[idx * 12 + h];
          }
        }
        sacc[nt] = wmma_bf16(a, kfr[nt], c);
      }
      // Row softmax: each row lives in one 16-lane half across the 4 tiles.
#pragma unroll
      for (int g = 0; g < 8; ++g) {
        float mx = fmaxf(fmaxf(sacc[0][g], sacc[1][g]), fmaxf(sacc[2][g], sacc[3][g]));
#pragma unroll
        for (int m = 8; m >= 1; m >>= 1) mx = fmaxf(mx, __shfl_xor(mx, m, 32));
        float e0 = __expf(sacc[0][g] - mx);
        float e1 = __expf(sacc[1][g] - mx);
        float e2 = __expf(sacc[2][g] - mx);
        float e3 = __expf(sacc[3][g] - mx);
        float sum = e0 + e1 + e2 + e3;
#pragma unroll
        for (int m = 8; m >= 1; m >>= 1) sum += __shfl_xor(sum, m, 32);
        const float inv = 1.0f / sum;
        sacc[0][g] = e0 * inv; sacc[1][g] = e1 * inv;
        sacc[2][g] = e2 * inv; sacc[3][g] = e3 * inv;
      }
      // Stage P (16x64 bf16) through wave-private LDS to re-enter A layout.
#pragma unroll
      for (int nt = 0; nt < 4; ++nt)
        store_d_bf16(pw, 0, nt * 16, LDP, lane, sacc[nt], 1.0f);
      v16bf pa0 = load_a_frag(pw, 0, 0,  LDP, lane);
      v16bf pa1 = load_a_frag(pw, 0, 32, LDP, lane);
#pragma unroll
      for (int dt = 0; dt < 2; ++dt) {
        v8f o = {0.f, 0.f, 0.f, 0.f, 0.f, 0.f, 0.f, 0.f};
        o = wmma_bf16(pa0, vfr[dt][0], o);
        o = wmma_bf16(pa1, vfr[dt][1], o);
        store_d_bf16(xn, mt * 16, h * 32 + dt * 16, LDX, lane, o, 1.0f);  // attn out reuses xn
      }
    }
  }
  __syncthreads();

  // ---------------- Phase 4: output projection + bias ----------------------
  // y[n][c] = sum_e attn[n][e] * w_out[c][e] + b_out[c]; w_out row-major == B^T.
  for (int ct = wave; ct < 24; ct += 8) {
    const int c0 = ct * 16;
    v16bf bfr[12];
#pragma unroll
    for (int k = 0; k < 12; ++k)
      bfr[k] = PRE ? load_bt_frag(wout_bf, c0, k * 32, 384, lane)
                   : load_bt_frag_f32(w_out, c0, k * 32, 384, lane);
    const float bb = b_out[c0 + (lane & 15)];
    for (int mt = 0; mt < 4; ++mt) {
      v8f acc = {0.f, 0.f, 0.f, 0.f, 0.f, 0.f, 0.f, 0.f};
#pragma unroll
      for (int k = 0; k < 12; ++k) {
        v16bf a = load_a_frag(xn, mt * 16, k * 32, LDX, lane);
        acc = wmma_bf16(a, bfr[k], acc);
      }
      const int c  = c0 + (lane & 15);
      const int hi = lane >> 4;
#pragma unroll
      for (int g = 0; g < 8; ++g) {
        const int row = mt * 16 + hi * 8 + g;
        out[xbase + row * 384 + c] = acc[g] + bb;
      }
    }
  }
}

extern "C" void kernel_launch(void* const* d_in, const int* in_sizes, int n_in,
                              void* d_out, int out_size, void* d_ws, size_t ws_size,
                              hipStream_t stream) {
  (void)in_sizes; (void)n_in; (void)out_size;
  const float* x          = (const float*)d_in[0];
  const float* ln_g       = (const float*)d_in[1];
  const float* ln_b       = (const float*)d_in[2];
  const float* w_qkv      = (const float*)d_in[3];
  const float* w_out      = (const float*)d_in[4];
  const float* b_out      = (const float*)d_in[5];
  const float* bias_table = (const float*)d_in[6];
  const int*   rel_idx    = (const int*)d_in[7];
  float* out = (float*)d_out;

  const bool use_ws = (ws_size >= (size_t)WS_REQ) && (d_ws != nullptr);

  // 212 KB of dynamic LDS (CDNA5 WGP supports up to 320 KB per workgroup).
  (void)hipFuncSetAttribute((const void*)swin_window_attn<true>,
                            hipFuncAttributeMaxDynamicSharedMemorySize, SMEM_BYTES);
  (void)hipFuncSetAttribute((const void*)swin_window_attn<false>,
                            hipFuncAttributeMaxDynamicSharedMemorySize, SMEM_BYTES);

  if (use_ws) {
    bf16*  wqkv_bf = (bf16*)((char*)d_ws + WS_OFF_WQKV);
    bf16*  wout_bf = (bf16*)((char*)d_ws + WS_OFF_WOUT);
    float* bias_pre = (float*)((char*)d_ws + WS_OFF_BIAS);
    const int total = NQKV + NOUT + NBIAS;
    prep_ws<<<(total + 255) / 256, 256, 0, stream>>>(
        w_qkv, w_out, bias_table, rel_idx, wqkv_bf, wout_bf, bias_pre);
    swin_window_attn<true><<<2048, 256, SMEM_BYTES, stream>>>(
        x, ln_g, ln_b, w_qkv, w_out, b_out, bias_table, rel_idx,
        wqkv_bf, wout_bf, bias_pre, out);
  } else {
    swin_window_attn<false><<<2048, 256, SMEM_BYTES, stream>>>(
        x, ln_g, ln_b, w_qkv, w_out, b_out, bias_table, rel_idx,
        nullptr, nullptr, nullptr, out);
  }
}
